// CNNDualMamba_41085657153685
// MI455X (gfx1250) — matrix operand order repaired and checked by reading, not compiled, not run on hardware
//
#include <hip/hip_runtime.h>
#include <hip/hip_bf16.h>
#include <math.h>

typedef _Float16 v16h __attribute__((ext_vector_type(16)));
typedef float    v8f  __attribute__((ext_vector_type(8)));
typedef float    v4f  __attribute__((ext_vector_type(4)));
typedef unsigned int u32x4 __attribute__((ext_vector_type(4)));
typedef int          i32x8 __attribute__((ext_vector_type(8)));
typedef int          i32x4 __attribute__((ext_vector_type(4)));

static inline int cdiv(int a, int b) { return (a + b - 1) / b; }

__device__ __forceinline__ float sigf(float x) { return 1.f / (1.f + __expf(-x)); }

// ---------------------------------------------------------------------------
// fragment loaders: fp32 -> f16 WMMA fragments via aligned b128 vector loads
// A fragment (16x32): lane<16 holds K 0-15, lane>=16 holds K 16-31
//   caller passes p = row_base + hi*16; loads p[0..15]
// B fragment (32x16): lane holds 16 contiguous K for its column, upper half +16
// ---------------------------------------------------------------------------
__device__ __forceinline__ v16h cvt_a_frag(const float* p)
{
    const v4f* q0 = (const v4f*)p;
    const v4f* q1 = (const v4f*)(p + 8);
    v4f x0 = q0[0], x1 = q0[1], x2 = q1[0], x3 = q1[1];
    v16h f;
    #pragma unroll
    for (int i = 0; i < 4; ++i) {
        f[i]      = (_Float16)x0[i];
        f[i + 4]  = (_Float16)x1[i];
        f[i + 8]  = (_Float16)x2[i];
        f[i + 12] = (_Float16)x3[i];
    }
    return f;
}

__device__ __forceinline__ v16h cvt_b_frag(const float* p)
{
    const v4f* q = (const v4f*)p;
    v4f y0 = q[0], y1 = q[1], y2 = q[2], y3 = q[3];
    v16h f;
    #pragma unroll
    for (int i = 0; i < 4; ++i) {
        f[i]      = (_Float16)y0[i];
        f[i + 4]  = (_Float16)y1[i];
        f[i + 8]  = (_Float16)y2[i];
        f[i + 12] = (_Float16)y3[i];
    }
    return f;
}

__device__ __forceinline__ float epi(float x, float bv, int act)
{
    x += bv;
    if (act == 1)      x = fmaxf(x, 0.f);
    else if (act == 2) x = (x > 20.f) ? x : log1pf(__expf(x));
    return x;
}

// ---------------------------------------------------------------------------
// TDM-staged WMMA GEMM: requires M%16==0, N%128==0, K%32==0.
// Block = 4 waves = 16M x 128N tile. The 16x32 A tile for each K chunk is
// DMA'd into LDS once per block by the Tensor Data Mover (double buffered,
// pad 4 dwords per 32-dword row -> LDS row stride 36), all 4 waves read A
// fragments from LDS; B streams from global as b128 loads.
// ---------------------------------------------------------------------------
#define ATILE_STRIDE 36

__global__ void k_gemm_wmma_tdm(const float* __restrict__ A, int lda,
                                const float* __restrict__ W, int ldw,
                                float* __restrict__ C, int ldc,
                                int M, int N, int K,
                                const float* __restrict__ bias, int act)
{
#if __has_builtin(__builtin_amdgcn_tensor_load_to_lds)
    __shared__ float atile[2][16 * ATILE_STRIDE];
    const int lane  = threadIdx.x;
    const int ty    = threadIdx.y;
    const int tpair = blockIdx.x * 4 + ty;
    const int tm    = blockIdx.y;
    const int l  = lane & 15;
    const int hi = lane >> 4;

    const float* wp0 = W + (size_t)(tpair * 32 + l) * ldw + hi * 16;
    const float* wp1 = wp0 + (size_t)16 * ldw;
    const float* abase = A + (size_t)(tm * 16) * lda;
    const int nk = K / 32;

    auto tdm_issue = [&](int ki, int buf) {
        unsigned long long ga = (unsigned long long)(size_t)(abase + (size_t)ki * 32);
        unsigned int lds = (unsigned int)(size_t)&atile[buf][0];  // addr[31:0] = LDS offset
        u32x4 g0;
        g0[0] = 1u;                                        // count=1 (valid descriptor)
        g0[1] = lds;                                       // lds_addr (bytes)
        g0[2] = (unsigned int)(ga & 0xFFFFFFFFu);          // global_addr[31:0]
        g0[3] = (unsigned int)((ga >> 32) & 0x1FFFFFFu)    // global_addr[56:32]
              | (2u << 30);                                // type=2 ("image")
        i32x8 g1;
        g1[0] = (2 << 16)                                  // data_size = 4B
              | (1 << 20)                                  // pad_enable
              | (4 << 22)                                  // pad_interval: 32 dwords
              | (3 << 25);                                 // pad_amount: 4 dwords
        g1[1] = (32 & 0xFFFF) << 16;                       // tensor_dim0 = 32 (lo16)
        g1[2] = (16 & 0xFFFF) << 16;                       // dim0 hi=0 | tensor_dim1 = 16 (lo16)
        g1[3] = (32 << 16);                                // dim1 hi=0 | tile_dim0 = 32
        g1[4] = 16;                                        // tile_dim1 = 16 | tile_dim2 = 0
        g1[5] = (int)(unsigned int)lda;                    // tensor_dim0_stride lo32 (elements)
        g1[6] = 0;                                         // stride hi16 | dim1_stride lo16
        g1[7] = 0;
        i32x4 gz  = {0, 0, 0, 0};                          // groups 2/3 zero (2D tensor)
        i32x8 gz8 = {0, 0, 0, 0, 0, 0, 0, 0};              // extra group (6-arg form), zero
        __builtin_amdgcn_tensor_load_to_lds(g0, g1, gz, gz, gz8, 0);
    };

    v8f acc0 = {}, acc1 = {};
    if (ty == 0) tdm_issue(0, 0);
    for (int ki = 0; ki < nk; ++ki) {
        if (ty == 0) __builtin_amdgcn_s_wait_tensorcnt(0);
        __syncthreads();                                   // publish buf[ki&1]
        if (ty == 0 && ki + 1 < nk) tdm_issue(ki + 1, (ki + 1) & 1);
        const float* ar = &atile[ki & 1][l * ATILE_STRIDE + hi * 16];
        v16h af = cvt_a_frag(ar);                          // ds_load_b128 path
        v16h b0 = cvt_b_frag(wp0 + (size_t)ki * 32);
        v16h b1 = cvt_b_frag(wp1 + (size_t)ki * 32);
        acc0 = __builtin_amdgcn_wmma_f32_16x16x32_f16(false, af, false, b0,
                                                      (short)0, acc0, false, false);
        acc1 = __builtin_amdgcn_wmma_f32_16x16x32_f16(false, af, false, b1,
                                                      (short)0, acc1, false, false);
    }

    const int col0 = tpair * 32 + l;
    const float bv0 = bias ? bias[col0]      : 0.f;
    const float bv1 = bias ? bias[col0 + 16] : 0.f;
    #pragma unroll
    for (int v = 0; v < 8; ++v) {
        int row = tm * 16 + hi * 8 + v;
        C[(size_t)row * ldc + col0]      = epi(acc0[v], bv0, act);
        C[(size_t)row * ldc + col0 + 16] = epi(acc1[v], bv1, act);
    }
#else
    // fallback: direct-load fast path (no TDM builtin on this toolchain)
    const int lane  = threadIdx.x;
    const int tpair = blockIdx.x * 4 + threadIdx.y;
    const int tm    = blockIdx.y;
    const int l  = lane & 15;
    const int hi = lane >> 4;
    const float* ap  = A + (size_t)(tm * 16 + l) * lda + hi * 16;
    const float* wp0 = W + (size_t)(tpair * 32 + l) * ldw + hi * 16;
    const float* wp1 = wp0 + (size_t)16 * ldw;
    v8f acc0 = {}, acc1 = {};
    for (int k0 = 0; k0 < K; k0 += 32) {
        v16h af = cvt_a_frag(ap + k0);
        v16h b0 = cvt_b_frag(wp0 + k0);
        v16h b1 = cvt_b_frag(wp1 + k0);
        acc0 = __builtin_amdgcn_wmma_f32_16x16x32_f16(false, af, false, b0,
                                                      (short)0, acc0, false, false);
        acc1 = __builtin_amdgcn_wmma_f32_16x16x32_f16(false, af, false, b1,
                                                      (short)0, acc1, false, false);
    }
    const int col0 = tpair * 32 + l;
    const float bv0 = bias ? bias[col0]      : 0.f;
    const float bv1 = bias ? bias[col0 + 16] : 0.f;
    #pragma unroll
    for (int v = 0; v < 8; ++v) {
        int row = tm * 16 + hi * 8 + v;
        C[(size_t)row * ldc + col0]      = epi(acc0[v], bv0, act);
        C[(size_t)row * ldc + col0 + 16] = epi(acc1[v], bv1, act);
    }
#endif
}

// ---------------------------------------------------------------------------
// FAST WMMA GEMM (direct loads): M%16==0, N%32==0, K%32==0
// ---------------------------------------------------------------------------
__global__ void k_gemm_wmma_fast(const float* __restrict__ A, int lda,
                                 const float* __restrict__ W, int ldw,
                                 float* __restrict__ C, int ldc,
                                 int M, int N, int K,
                                 const float* __restrict__ bias, int act)
{
    const int lane  = threadIdx.x;
    const int tpair = blockIdx.x * blockDim.y + threadIdx.y;
    const int tm    = blockIdx.y;
    if (tpair * 32 >= N) return;                              // uniform per wave
    const int l  = lane & 15;
    const int hi = lane >> 4;

    const float* ap  = A + (size_t)(tm * 16 + l) * lda + hi * 16;
    const float* wp0 = W + (size_t)(tpair * 32 + l) * ldw + hi * 16;
    const float* wp1 = wp0 + (size_t)16 * ldw;

    v8f acc0 = {}, acc1 = {};
    for (int k0 = 0; k0 < K; k0 += 32) {
        __builtin_prefetch(ap  + k0 + 256, 0, 3);             // global_prefetch_b8
        __builtin_prefetch(wp0 + k0 + 256, 0, 3);
        v16h af = cvt_a_frag(ap + k0);
        v16h b0 = cvt_b_frag(wp0 + k0);
        v16h b1 = cvt_b_frag(wp1 + k0);
        acc0 = __builtin_amdgcn_wmma_f32_16x16x32_f16(false, af, false, b0,
                                                      (short)0, acc0, false, false);
        acc1 = __builtin_amdgcn_wmma_f32_16x16x32_f16(false, af, false, b1,
                                                      (short)0, acc1, false, false);
    }

    const int col0 = tpair * 32 + l;
    const float bv0 = bias ? bias[col0]      : 0.f;
    const float bv1 = bias ? bias[col0 + 16] : 0.f;
    #pragma unroll
    for (int v = 0; v < 8; ++v) {
        int row = tm * 16 + hi * 8 + v;
        C[(size_t)row * ldc + col0]      = epi(acc0[v], bv0, act);
        C[(size_t)row * ldc + col0 + 16] = epi(acc1[v], bv1, act);
    }
}

// ---------------------------------------------------------------------------
// EDGE WMMA GEMM: arbitrary M,N,K
// ---------------------------------------------------------------------------
__global__ void k_gemm_wmma_edge(const float* __restrict__ A, int lda,
                                 const float* __restrict__ W, int ldw,
                                 float* __restrict__ C, int ldc,
                                 int M, int N, int K,
                                 const float* __restrict__ bias, int act)
{
    const int lane = threadIdx.x;
    const int tn = blockIdx.x * blockDim.y + threadIdx.y;
    const int tm = blockIdx.y;
    if (tn * 16 >= N) return;
    const int l  = lane & 15;
    const int hi = lane >> 4;
    const int mrow = tm * 16 + l;
    const int ncol = tn * 16 + l;
    const bool mok = mrow < M;
    const bool nok = ncol < N;
    const float* ap = A + (size_t)(mok ? mrow : 0) * lda + hi * 16;
    const float* wp = W + (size_t)(nok ? ncol : 0) * ldw + hi * 16;

    v8f acc = {};
    const int Kmain = K & ~31;
    int k0 = 0;
    for (; k0 < Kmain; k0 += 32) {
        v16h af = {}, bf = {};
        if (mok) af = cvt_a_frag(ap + k0);
        if (nok) bf = cvt_b_frag(wp + k0);
        acc = __builtin_amdgcn_wmma_f32_16x16x32_f16(false, af, false, bf,
                                                     (short)0, acc, false, false);
    }
    if (k0 < K) {
        v16h af = {}, bf = {};
        if (mok) {
            #pragma unroll
            for (int i = 0; i < 8; ++i) {
                int ka = k0 + hi * 16 + i;
                if (ka < K)      af[i]     = (_Float16)ap[k0 + i];
                if (ka + 8 < K)  af[i + 8] = (_Float16)ap[k0 + i + 8];
            }
        }
        if (nok) {
            #pragma unroll
            for (int i = 0; i < 16; ++i) {
                int kb = k0 + hi * 16 + i;
                if (kb < K) bf[i] = (_Float16)wp[k0 + i];
            }
        }
        acc = __builtin_amdgcn_wmma_f32_16x16x32_f16(false, af, false, bf,
                                                     (short)0, acc, false, false);
    }

    if (!nok) return;
    float bv = bias ? bias[ncol] : 0.f;
    #pragma unroll
    for (int v = 0; v < 8; ++v) {
        int row = tm * 16 + hi * 8 + v;
        if (row < M)
            C[(size_t)row * ldc + ncol] = epi(acc[v], bv, act);
    }
}

static void gemm(const float* A, int lda, const float* W, int ldw,
                 float* C, int ldc, int M, int N, int K,
                 const float* bias, int act, hipStream_t s)
{
    if ((M % 16 == 0) && (K % 32 == 0) && (N % 128 == 0)) {
        dim3 blk(32, 4);
        dim3 grd(N / 128, M / 16);
        k_gemm_wmma_tdm<<<grd, blk, 0, s>>>(A, lda, W, ldw, C, ldc, M, N, K, bias, act);
    } else if ((M % 16 == 0) && (N % 32 == 0) && (K % 32 == 0)) {
        dim3 blk(32, 4);
        dim3 grd(cdiv(N / 32, 4), M / 16);
        k_gemm_wmma_fast<<<grd, blk, 0, s>>>(A, lda, W, ldw, C, ldc, M, N, K, bias, act);
    } else {
        dim3 blk(32, 4);
        dim3 grd(cdiv(cdiv(N, 16), 4), cdiv(M, 16));
        k_gemm_wmma_edge<<<grd, blk, 0, s>>>(A, lda, W, ldw, C, ldc, M, N, K, bias, act);
    }
}

// ---------------------------------------------------------------------------
// conv3x3 (pad 1) + bias + BN + ReLU + 2x2 maxpool
// ---------------------------------------------------------------------------
__global__ void k_conv_bn_relu_pool(const float* __restrict__ in,
                                    const float* __restrict__ w,
                                    const float* __restrict__ bias,
                                    const float* __restrict__ gg,
                                    const float* __restrict__ bb,
                                    const float* __restrict__ mm,
                                    const float* __restrict__ vv,
                                    float* __restrict__ out,
                                    int B, int Ci, int H, int Wd, int Co)
{
    int Hp = H / 2, Wp = Wd / 2;
    int idx = blockIdx.x * blockDim.x + threadIdx.x;
    int total = B * Co * Hp * Wp;
    if (idx >= total) return;
    int pw = idx % Wp; int t = idx / Wp;
    int ph = t % Hp;   t /= Hp;
    int co = t % Co;   int b = t / Co;

    float scale = gg[co] * rsqrtf(vv[co] + 1e-5f);
    float shift = bb[co] - mm[co] * scale;
    float bsv   = bias[co];
    const float* wco = w + (size_t)co * Ci * 9;

    float best = -1e30f;
    for (int dy = 0; dy < 2; ++dy)
    for (int dx = 0; dx < 2; ++dx) {
        int h0 = ph * 2 + dy, w0 = pw * 2 + dx;
        float acc = bsv;
        for (int ci = 0; ci < Ci; ++ci) {
            const float* ip = in + ((size_t)(b * Ci + ci) * H) * Wd;
            const float* wk = wco + ci * 9;
            #pragma unroll
            for (int kh = 0; kh < 3; ++kh) {
                int hh = h0 + kh - 1;
                if (hh < 0 || hh >= H) continue;
                #pragma unroll
                for (int kw = 0; kw < 3; ++kw) {
                    int ww = w0 + kw - 1;
                    if (ww < 0 || ww >= Wd) continue;
                    acc += wk[kh * 3 + kw] * ip[(size_t)hh * Wd + ww];
                }
            }
        }
        float bnv = fmaxf(acc * scale + shift, 0.f);
        best = fmaxf(best, bnv);
    }
    out[idx] = best;
}

// ---------------------------------------------------------------------------
// CBAM helpers
// ---------------------------------------------------------------------------
__global__ void k_chanmean(const float* __restrict__ x, float* __restrict__ mean,
                           int BC, int HW)
{
    int i = blockIdx.x * blockDim.x + threadIdx.x;
    if (i >= BC) return;
    const float* p = x + (size_t)i * HW;
    float s = 0.f;
    for (int j = 0; j < HW; ++j) s += p[j];
    mean[i] = s / (float)HW;
}

__global__ void k_camlp(const float* __restrict__ mean, const float* __restrict__ w1,
                        const float* __restrict__ w2, float* __restrict__ ca,
                        int C, int Hd)
{
    int b = blockIdx.x, t = threadIdx.x;   // 128 threads
    __shared__ float sm[128];
    __shared__ float hid[8];
    sm[t] = mean[b * C + t];
    __syncthreads();
    if (t < Hd) {
        float s = 0.f;
        for (int c = 0; c < C; ++c) s += w1[t * C + c] * sm[c];
        hid[t] = fmaxf(s, 0.f);
    }
    __syncthreads();
    float s = 0.f;
    for (int j = 0; j < Hd; ++j) s += w2[t * Hd + j] * hid[j];
    ca[b * C + t] = sigf(s);
}

__global__ void k_spstats(const float* __restrict__ h, const float* __restrict__ ca,
                          float* __restrict__ avg, float* __restrict__ mx,
                          int B, int C, int HW)
{
    int i = blockIdx.x * blockDim.x + threadIdx.x;
    if (i >= B * HW) return;
    int b = i / HW, s = i % HW;
    float sum = 0.f, m = -1e30f;
    for (int c = 0; c < C; ++c) {
        float v = h[((size_t)(b * C + c)) * HW + s] * ca[b * C + c];
        sum += v; m = fmaxf(m, v);
    }
    avg[i] = sum / (float)C;
    mx[i]  = m;
}

__global__ void k_sa(const float* __restrict__ avg, const float* __restrict__ mx,
                     const float* __restrict__ ws, float* __restrict__ sa,
                     int B, int HH, int WW)
{
    int i = blockIdx.x * blockDim.x + threadIdx.x;
    if (i >= B * HH * WW) return;
    int w = i % WW; int t = i / WW; int h = t % HH; int b = t / HH;
    float acc = 0.f;
    for (int ch = 0; ch < 2; ++ch) {
        const float* pl = (ch ? mx : avg) + (size_t)b * HH * WW;
        #pragma unroll
        for (int kh = 0; kh < 3; ++kh) {
            int hh = h + kh - 1;
            if (hh < 0 || hh >= HH) continue;
            #pragma unroll
            for (int kw = 0; kw < 3; ++kw) {
                int wc = w + kw - 1;
                if (wc < 0 || wc >= WW) continue;
                acc += ws[ch * 9 + kh * 3 + kw] * pl[hh * WW + wc];
            }
        }
    }
    sa[i] = sigf(acc);
}

__global__ void k_apply(const float* __restrict__ h3, const float* __restrict__ ca,
                        const float* __restrict__ sa,
                        float* __restrict__ xt, float* __restrict__ xf,
                        int B, int C, int HH, int WW)
{
    int i = blockIdx.x * blockDim.x + threadIdx.x;
    int total = B * C * HH * WW;
    if (i >= total) return;
    int w = i % WW; int t = i / WW; int h = t % HH; t /= HH;
    int c = t % C;  int b = t / C;
    float v = h3[i] * ca[b * C + c] * sa[(b * HH + h) * WW + w];
    xt[((size_t)(b * WW + w)) * (C * HH) + c * HH + h] = v;  // (B, WW, C*HH)
    xf[((size_t)(b * HH + h)) * (C * WW) + c * WW + w] = v;  // (B, HH, C*WW)
}

// ---------------------------------------------------------------------------
// depthwise causal conv1d (k=4, left pad 3) + bias + SiLU
// ---------------------------------------------------------------------------
__global__ void k_dwconv_silu(const float* __restrict__ xz, const float* __restrict__ cw,
                              const float* __restrict__ cb, float* __restrict__ xc,
                              int B, int L, int di)
{
    int i = blockIdx.x * blockDim.x + threadIdx.x;
    if (i >= B * L * di) return;
    int d = i % di; int t = i / di; int l = t % L; int b = t / L;
    float s = cb[d];
    #pragma unroll
    for (int k = 0; k < 4; ++k) {
        int ls = l - 3 + k;
        if (ls >= 0) s += cw[d * 4 + k] * xz[((size_t)(b * L + ls)) * (2 * di) + d];
    }
    xc[i] = s * sigf(s);
}

// ---------------------------------------------------------------------------
// selective scan, one thread per (b,d); fused +u*D and *silu(z)
// ---------------------------------------------------------------------------
__global__ void k_selscan(const float* __restrict__ xc, const float* __restrict__ dt,
                          const float* __restrict__ xdbl, int ldx, int roff,
                          const float* __restrict__ A_log, const float* __restrict__ Dp,
                          const float* __restrict__ xz,
                          float* __restrict__ yz, int B, int L, int di)
{
    int i = blockIdx.x * blockDim.x + threadIdx.x;
    if (i >= B * di) return;
    int d = i % di, b = i / di;
    float a[16], h[16];
    #pragma unroll
    for (int n = 0; n < 16; ++n) { a[n] = -__expf(A_log[(size_t)d * 16 + n]); h[n] = 0.f; }
    float Dd = Dp[d];
    for (int l = 0; l < L; ++l) {
        size_t row = (size_t)b * L + l;
        float dtv = dt[row * di + d];
        float u   = xc[row * di + d];
        const float* Bm = xdbl + row * ldx + roff;
        const float* Cm = Bm + 16;
        float du = dtv * u;
        float y = 0.f;
        #pragma unroll
        for (int n = 0; n < 16; ++n) {
            h[n] = __expf(dtv * a[n]) * h[n] + du * Bm[n];
            y += h[n] * Cm[n];
        }
        y += u * Dd;
        float z = xz[row * (size_t)(2 * di) + di + d];
        yz[row * di + d] = y * (z * sigf(z));
    }
}

// ---------------------------------------------------------------------------
// row LayerNorm (in place), one block per row; then mean over L
// ---------------------------------------------------------------------------
__global__ void k_layernorm(float* __restrict__ X, const float* __restrict__ g,
                            const float* __restrict__ b, int dm)
{
    int r = blockIdx.x, t = threadIdx.x;
    __shared__ float red[256];
    __shared__ float red2[256];
    float* row = X + (size_t)r * dm;
    float s = 0.f, s2 = 0.f;
    for (int i = t; i < dm; i += 256) { float v = row[i]; s += v; s2 += v * v; }
    red[t] = s; red2[t] = s2;
    __syncthreads();
    for (int off = 128; off > 0; off >>= 1) {
        if (t < off) { red[t] += red[t + off]; red2[t] += red2[t + off]; }
        __syncthreads();
    }
    float mu  = red[0] / dm;
    float var = red2[0] / dm - mu * mu;
    float rstd = rsqrtf(var + 1e-5f);
    for (int i = t; i < dm; i += 256)
        row[i] = (row[i] - mu) * rstd * g[i] + b[i];
}

__global__ void k_meanL(const float* __restrict__ Y, float* __restrict__ z,
                        int B, int L, int dm, int zld, int zoff)
{
    int i = blockIdx.x * blockDim.x + threadIdx.x;
    if (i >= B * dm) return;
    int d = i % dm, b = i / dm;
    float s = 0.f;
    for (int l = 0; l < L; ++l) s += Y[((size_t)(b * L + l)) * dm + d];
    z[(size_t)b * zld + zoff + d] = s / (float)L;
}

// ---------------------------------------------------------------------------
// host
// ---------------------------------------------------------------------------
extern "C" void kernel_launch(void* const* d_in, const int* in_sizes, int n_in,
                              void* d_out, int out_size, void* d_ws, size_t ws_size,
                              hipStream_t stream)
{
    (void)in_sizes; (void)n_in; (void)out_size; (void)ws_size;
    auto F = [&](int i) { return (const float*)d_in[i]; };

    const float* x = F(0);
    struct MP { const float *in_w, *conv_w, *conv_b, *xproj_w, *dtproj_w, *dtproj_b, *A_log, *D, *out_w; };
    MP mt = { F(22), F(23), F(24), F(25), F(26), F(27), F(28), F(29), F(30) };
    MP mf = { F(31), F(32), F(33), F(34), F(35), F(36), F(37), F(38), F(39) };
    const float *nt_g = F(40), *nt_b = F(41), *nf_g = F(42), *nf_b = F(43);
    const float *red_w = F(44), *red_b = F(45);
    const float *fc1_w = F(46), *fc1_b = F(47);
    const float *fc2_w = F(48), *fc2_b = F(49);

    // ---- workspace layout (floats). Mamba scratch overlaps retired conv bufs.
    float* Wf = (float*)d_ws;
    const size_t MS = 17061376;                // xz+xc+xdbl+dt+yz+yo
    float* p_xz   = Wf;                        // 6,160,384
    float* p_xc   = p_xz + 6160384;            // 3,080,192
    float* p_xdbl = p_xc + 3080192;            //   120,320
    float* p_dt   = p_xdbl + 120320;           // 3,080,192
    float* p_yz   = p_dt + 3080192;            // 3,080,192
    float* p_yo   = p_yz + 3080192;            // 1,540,096
    float* p_c1 = Wf;                          // conv scratch reuses front
    float* p_c2 = p_c1 + 6160384;
    size_t o = MS;
    float* p_c3     = Wf + o; o += 16 * 128 * 16 * 47;
    float* p_camean = Wf + o; o += 16 * 128;
    float* p_ca     = Wf + o; o += 16 * 128;
    float* p_avg    = Wf + o; o += 16 * 16 * 47;
    float* p_mx     = Wf + o; o += 16 * 16 * 47;
    float* p_sa     = Wf + o; o += 16 * 16 * 47;
    float* p_xt     = Wf + o; o += 16 * 47 * 2048;
    float* p_xf     = Wf + o; o += 16 * 16 * 6016;
    float* p_zcat   = Wf + o; o += 16 * 8064;
    float* p_r1     = Wf + o; o += 16 * 1024;
    float* p_r2     = Wf + o; o += 16 * 256;

    const int TB = 256;

    // ---- CNN front-end
    {
        int t1 = 16 * 32 * 64 * 188;
        k_conv_bn_relu_pool<<<cdiv(t1, TB), TB, 0, stream>>>(
            x, F(1), F(2), F(3), F(4), F(5), F(6), p_c1, 16, 1, 128, 376, 32);
        int t2 = 16 * 64 * 32 * 94;
        k_conv_bn_relu_pool<<<cdiv(t2, TB), TB, 0, stream>>>(
            p_c1, F(7), F(8), F(9), F(10), F(11), F(12), p_c2, 16, 32, 64, 188, 64);
        int t3 = 16 * 128 * 16 * 47;
        k_conv_bn_relu_pool<<<cdiv(t3, TB), TB, 0, stream>>>(
            p_c2, F(13), F(14), F(15), F(16), F(17), F(18), p_c3, 16, 64, 32, 94, 128);
    }

    // ---- CBAM (16,128,16,47)
    {
        int HW = 16 * 47;
        k_chanmean<<<cdiv(2048, TB), TB, 0, stream>>>(p_c3, p_camean, 16 * 128, HW);
        k_camlp<<<16, 128, 0, stream>>>(p_camean, F(19), F(20), p_ca, 128, 8);
        k_spstats<<<cdiv(16 * HW, TB), TB, 0, stream>>>(p_c3, p_ca, p_avg, p_mx, 16, 128, HW);
        k_sa<<<cdiv(16 * HW, TB), TB, 0, stream>>>(p_avg, p_mx, F(21), p_sa, 16, 16, 47);
        int tot = 16 * 128 * 16 * 47;
        k_apply<<<cdiv(tot, TB), TB, 0, stream>>>(p_c3, p_ca, p_sa, p_xt, p_xf, 16, 128, 16, 47);
    }

    // ---- dual Mamba branches (identical scratch footprints)
    auto run_mamba = [&](const float* X, int L, int dm, const MP& mp,
                         const float* lng, const float* lnb, int zoff) {
        const int B = 16, BL = B * L;
        const int di = 2 * dm, r = (dm + 15) / 16, ldx = r + 32;
        gemm(X, dm, mp.in_w, dm, p_xz, 2 * di, BL, 2 * di, dm, nullptr, 0, stream);
        k_dwconv_silu<<<cdiv(BL * di, TB), TB, 0, stream>>>(p_xz, mp.conv_w, mp.conv_b, p_xc, B, L, di);
        gemm(p_xc, di, mp.xproj_w, di, p_xdbl, ldx, BL, ldx, di, nullptr, 0, stream);
        gemm(p_xdbl, ldx, mp.dtproj_w, r, p_dt, di, BL, di, r, mp.dtproj_b, 2, stream);
        k_selscan<<<cdiv(B * di, TB), TB, 0, stream>>>(p_xc, p_dt, p_xdbl, ldx, r,
                                                       mp.A_log, mp.D, p_xz, p_yz, B, L, di);
        gemm(p_yz, di, mp.out_w, di, p_yo, dm, BL, dm, di, nullptr, 0, stream);
        k_layernorm<<<BL, 256, 0, stream>>>(p_yo, lng, lnb, dm);
        k_meanL<<<cdiv(B * dm, TB), TB, 0, stream>>>(p_yo, p_zcat, B, L, dm, 8064, zoff);
    };
    run_mamba(p_xt, 47, 2048, mt, nt_g, nt_b, 0);
    run_mamba(p_xf, 16, 6016, mf, nf_g, nf_b, 2048);

    // ---- head: reduce -> relu fc1 -> fc2
    gemm(p_zcat, 8064, red_w, 8064, p_r1, 1024, 16, 1024, 8064, red_b, 0, stream);
    gemm(p_r1, 1024, fc1_w, 1024, p_r2, 256, 16, 256, 1024, fc1_b, 1, stream);
    gemm(p_r2, 256, fc2_w, 256, (float*)d_out, 36, 16, 36, 256, fc2_b, 0, stream);
}
